// SiameseImage_55843164783347
// MI455X (gfx1250) — matrix operand order repaired and checked by reading, not compiled, not run on hardware
//
#include <hip/hip_runtime.h>

// ---------------------------------------------------------------------------
// CDNA5 (gfx1250) wave32 WMMA types
// ---------------------------------------------------------------------------
typedef __attribute__((ext_vector_type(4)))  unsigned int v4u;
typedef __attribute__((ext_vector_type(8)))  float        v8f;
typedef __attribute__((ext_vector_type(8)))  unsigned int v8u;
typedef __attribute__((ext_vector_type(16))) __bf16       v16bf;

union BF16x16 { v8u u; v16bf h; };

__device__ __forceinline__ unsigned short f2bf(float f) {
    unsigned int u = __float_as_uint(f);
    unsigned int r = u + 0x7fffu + ((u >> 16) & 1u);   // round-to-nearest-even
    return (unsigned short)(r >> 16);
}

// ---------------------------------------------------------------------------
// CDNA5 async global->LDS copy (ASYNCcnt path). First operand = LDS byte
// address VGPR, second = 64-bit global address, no SADDR. Generic LDS
// pointers carry the DS byte offset in their low 32 bits.
// ---------------------------------------------------------------------------
__device__ __forceinline__ void async_b128(void* lds_dst, const void* gsrc) {
    unsigned lds = (unsigned)(size_t)lds_dst;
    asm volatile("global_load_async_to_lds_b128 %0, %1, off"
                 :: "v"(lds), "v"(gsrc) : "memory");
}
__device__ __forceinline__ void async_wait0() {
    asm volatile("s_wait_asynccnt 0" ::: "memory");
}

// ---------------------------------------------------------------------------
// Weight prep: conv_ws[1..5] (64,64,3,3) f32 -> bf16 [conv][tap][oc][ic]
// ---------------------------------------------------------------------------
__global__ void k_prep_convw(const float* __restrict__ w1, const float* __restrict__ w2,
                             const float* __restrict__ w3, const float* __restrict__ w4,
                             const float* __restrict__ w5, unsigned short* __restrict__ wtb)
{
    int e = blockIdx.x * blockDim.x + threadIdx.x;
    if (e >= 5 * 36864) return;
    int c = e / 36864, r = e % 36864;
    int ic = r & 63; int t = r >> 6; int oc = t & 63; int tap = t >> 6;
    const float* w = (c == 0) ? w1 : (c == 1) ? w2 : (c == 2) ? w3 : (c == 3) ? w4 : w5;
    wtb[e] = f2bf(w[(oc * 64 + ic) * 9 + tap]);
}

// lin1_w (4096,64) f32 -> bf16 [oc][k]
__global__ void k_prep_lin1(const float* __restrict__ w, unsigned short* __restrict__ wb)
{
    int e = blockIdx.x * blockDim.x + threadIdx.x;
    if (e >= 64 * 4096) return;
    int k = e & 4095, oc = e >> 12;
    wb[e] = f2bf(w[k * 64 + oc]);
}

// ---------------------------------------------------------------------------
// conv0 (3->64, 64x64) fused with maxpool2 + relu, writing bf16 padded NHWC
// pad layout: [img][(34)(34)][64ic], zero border
// ---------------------------------------------------------------------------
__global__ void k_conv0_pool_pad(const float* __restrict__ x,   // [128][3][64][64]
                                 const float* __restrict__ w0,  // [64][3][3][3]
                                 const float* __restrict__ b0,  // [64]
                                 unsigned short* __restrict__ pad)
{
    __shared__ __align__(16) float ws[1728];
    __shared__ __align__(16) float bs[64];
    for (int i = threadIdx.x; i < 432; i += blockDim.x)
        async_b128(&ws[i * 4], &w0[i * 4]);                 // 1728 floats
    if (threadIdx.x < 16)
        async_b128(&bs[threadIdx.x * 4], &b0[threadIdx.x * 4]);
    async_wait0();
    __syncthreads();

    int e = blockIdx.x * blockDim.x + threadIdx.x;           // 128*34*34
    if (e >= 128 * 34 * 34) return;
    int xx = e % 34; int t = e / 34; int yy = t % 34; int img = t / 34;
    unsigned short* po = pad + (size_t)e * 64;
    if (yy == 0 || yy == 33 || xx == 0 || xx == 33) {
        for (int oc = 0; oc < 64; ++oc) po[oc] = 0;
        return;
    }
    int yp = yy - 1, xp = xx - 1;
    float patch[3][4][4];
    const float* xi = x + (size_t)img * 3 * 4096;
    for (int c = 0; c < 3; ++c)
        for (int r = 0; r < 4; ++r) {
            int y = 2 * yp - 1 + r;
            for (int q = 0; q < 4; ++q) {
                int xc = 2 * xp - 1 + q;
                patch[c][r][q] = (y >= 0 && y < 64 && xc >= 0 && xc < 64)
                                     ? xi[c * 4096 + y * 64 + xc] : 0.f;
            }
        }
    for (int oc = 0; oc < 64; ++oc) {
        float a00 = 0, a01 = 0, a10 = 0, a11 = 0;
        const float* wk = ws + oc * 27;
        #pragma unroll
        for (int c = 0; c < 3; ++c)
            #pragma unroll
            for (int ky = 0; ky < 3; ++ky)
                #pragma unroll
                for (int kx = 0; kx < 3; ++kx) {
                    float wv = wk[c * 9 + ky * 3 + kx];
                    a00 += patch[c][ky    ][kx    ] * wv;
                    a01 += patch[c][ky    ][kx + 1] * wv;
                    a10 += patch[c][ky + 1][kx    ] * wv;
                    a11 += patch[c][ky + 1][kx + 1] * wv;
                }
        float m = fmaxf(fmaxf(a00, a01), fmaxf(a10, a11)) + bs[oc];
        po[oc] = f2bf(fmaxf(m, 0.f));
    }
}

// ---------------------------------------------------------------------------
// 64->64 3x3 conv as implicit GEMM: 9 shifted taps x 2 K-steps of WMMA bf16.
// Block = 128 threads (4 waves); wave w computes 16-pixel x 16-oc tile.
// Per-lane operand dwords are contiguous -> two b128 loads per fragment:
//   A: dwords [4h..4h+3] and [8+4h..11+4h]   (h = lane>>4)
//   B: dwords [4h..4h+3] and [8+4h..11+4h]
// C/D (16x16 f32): vgpr r -> row r+8*half, col = lane&15
// Spatial size S = 1<<logS (32/16/8) -> shift/mask instead of int division.
// ---------------------------------------------------------------------------
__global__ void k_conv_wmma(const unsigned short* __restrict__ pad, // [img][(S+2)^2][64] bf16
                            const unsigned short* __restrict__ wtb, // [9][64oc][64ic] bf16
                            const float* __restrict__ bias,
                            float* __restrict__ out,                // [img][S*S][64] f32
                            int logS)
{
    const unsigned int* padu = (const unsigned int*)pad;
    const unsigned int* wu   = (const unsigned int*)wtb;
    int S = 1 << logS;
    int tile = blockIdx.x, img = blockIdx.y;
    int lane = threadIdx.x & 31;
    int ocb  = (threadIdx.x >> 5) * 16;
    int half = lane >> 4, nm = lane & 15;
    int p = tile * 16 + nm;               // A-row pixel for this lane
    int y = p >> logS, x = p & (S - 1);
    int Wp = S + 2;
    int imgbase = img * (S + 2) * Wp * 32;   // uint (=2 bf16) units
    v8f acc = {0.f, 0.f, 0.f, 0.f, 0.f, 0.f, 0.f, 0.f};

    #pragma unroll
    for (int tap = 0; tap < 9; ++tap) {
        int dy = tap / 3, dx = tap % 3;
        int abase = imgbase + ((y + dy) * Wp + (x + dx)) * 32;
        int brow  = (tap * 64 + ocb + nm) * 32 + 4 * half;
        #pragma unroll
        for (int s = 0; s < 2; ++s) {
            const unsigned int* ap = padu + abase + s * 16 + 4 * half;
            const unsigned int* bp = wu + brow + s * 16;
            v4u a0 = *(const v4u*)(ap);
            v4u a1 = *(const v4u*)(ap + 8);
            v4u b0 = *(const v4u*)(bp);
            v4u b1 = *(const v4u*)(bp + 8);
            BF16x16 a, b;
            a.u = __builtin_shufflevector(a0, a1, 0, 1, 2, 3, 4, 5, 6, 7);
            b.u = __builtin_shufflevector(b0, b1, 0, 1, 2, 3, 4, 5, 6, 7);
            acc = __builtin_amdgcn_wmma_f32_16x16x32_bf16(false, a.h, false, b.h,
                                                          (short)0, acc, false, false);
        }
    }
    float bi = bias[ocb + nm];
    int obase = ((img << (2 * logS)) + tile * 16) * 64 + ocb + nm;
    #pragma unroll
    for (int r = 0; r < 8; ++r)
        out[obase + (r + 8 * half) * 64] = acc[r] + bi;
}

// ---------------------------------------------------------------------------
// maxpool2 + relu + write bf16 padded NHWC (covers full padded extent -> zeros border)
// ---------------------------------------------------------------------------
__global__ void k_pool_relu_pad(const float* __restrict__ in, // [128][H*W][64]
                                unsigned short* __restrict__ pad, int H, int W)
{
    int Ho = H >> 1, Wo = W >> 1, Hp = Ho + 2, Wp = Wo + 2;
    int e = blockIdx.x * blockDim.x + threadIdx.x;
    if (e >= 128 * Hp * Wp * 64) return;
    int ic = e & 63; int t = e >> 6;
    int xx = t % Wp; t /= Wp; int yy = t % Hp; int img = t / Hp;
    unsigned short r;
    if (yy == 0 || yy == Hp - 1 || xx == 0 || xx == Wp - 1) r = 0;
    else {
        int y = (yy - 1) * 2, x = (xx - 1) * 2;
        const float* b = in + ((size_t)img * H * W + y * W + x) * 64 + ic;
        float m = fmaxf(fmaxf(b[0], b[64]), fmaxf(b[W * 64], b[W * 64 + 64]));
        r = f2bf(fmaxf(m, 0.f));
    }
    pad[e] = r;
}

// ---------------------------------------------------------------------------
// LayerNorm pass 1: per-image mean / rstd over npix*64
// ---------------------------------------------------------------------------
__global__ void k_ln_stats(const float* __restrict__ in, float* __restrict__ mv, int npix)
{
    __shared__ float ss[256], sq[256];
    int img = blockIdx.x;
    const float* b = in + (size_t)img * npix * 64;
    int n = npix * 64;
    float s = 0.f, q = 0.f;
    for (int i = threadIdx.x; i < n; i += 256) { float v = b[i]; s += v; q += v * v; }
    ss[threadIdx.x] = s; sq[threadIdx.x] = q;
    __syncthreads();
    for (int d = 128; d > 0; d >>= 1) {
        if (threadIdx.x < d) { ss[threadIdx.x] += ss[threadIdx.x + d];
                               sq[threadIdx.x] += sq[threadIdx.x + d]; }
        __syncthreads();
    }
    if (threadIdx.x == 0) {
        float mean = ss[0] / n;
        float var  = sq[0] / n - mean * mean;
        mv[img * 2] = mean;
        mv[img * 2 + 1] = rsqrtf(var + 1e-5f);
    }
}

// LayerNorm pass 2 + relu + bf16 padded write (g,b are CHW per reference)
__global__ void k_ln_relu_pad(const float* __restrict__ in, const float* __restrict__ mv,
                              const float* __restrict__ g,  const float* __restrict__ bb,
                              unsigned short* __restrict__ pad, int H, int W)
{
    int Hp = H + 2, Wp = W + 2;
    int e = blockIdx.x * blockDim.x + threadIdx.x;
    if (e >= 128 * Hp * Wp * 64) return;
    int ic = e & 63; int t = e >> 6;
    int xx = t % Wp; t /= Wp; int yy = t % Hp; int img = t / Hp;
    unsigned short r;
    if (yy == 0 || yy == Hp - 1 || xx == 0 || xx == Wp - 1) r = 0;
    else {
        int y = yy - 1, x = xx - 1;
        float v = in[((size_t)img * H * W + y * W + x) * 64 + ic];
        int gi = (ic * H + y) * W + x;
        r = f2bf(fmaxf((v - mv[img * 2]) * mv[img * 2 + 1] * g[gi] + bb[gi], 0.f));
    }
    pad[e] = r;
}

// LayerNorm pass 2 (final layer, 8x8) -> flat CHW bf16 features for lin1
__global__ void k_ln_relu_flat(const float* __restrict__ in, const float* __restrict__ mv,
                               const float* __restrict__ g,  const float* __restrict__ bb,
                               unsigned short* __restrict__ feat) // [128][4096] CHW
{
    int e = blockIdx.x * blockDim.x + threadIdx.x;
    if (e >= 128 * 4096) return;
    int k = e & 4095, img = e >> 12;
    int ic = k >> 6, pix = k & 63;
    float v = in[((size_t)img * 64 + pix) * 64 + ic];
    feat[e] = f2bf(fmaxf((v - mv[img * 2]) * mv[img * 2 + 1] * g[k] + bb[k], 0.f));
}

// ---------------------------------------------------------------------------
// lin1: [128 x 4096] @ [4096 x 64] + bias, relu — WMMA bf16, K loop = 128 steps
// ---------------------------------------------------------------------------
__global__ void k_lin1_wmma(const unsigned short* __restrict__ feat, // [128][4096] bf16
                            const unsigned short* __restrict__ wb,   // [64oc][4096] bf16
                            const float* __restrict__ bias,
                            float* __restrict__ h1)                  // [128][64] f32
{
    const unsigned int* fu = (const unsigned int*)feat;
    const unsigned int* wu = (const unsigned int*)wb;
    int tile = blockIdx.x;                 // 8 tiles of 16 rows
    int lane = threadIdx.x & 31;
    int ocb  = (threadIdx.x >> 5) * 16;
    int half = lane >> 4, nm = lane & 15;
    int arow = (tile * 16 + nm) * 2048 + 4 * half;   // uint units
    int brow = (ocb + nm) * 2048 + 4 * half;
    v8f acc = {0.f, 0.f, 0.f, 0.f, 0.f, 0.f, 0.f, 0.f};
    for (int s = 0; s < 128; ++s) {
        const unsigned int* ap = fu + arow + s * 16;
        const unsigned int* bp = wu + brow + s * 16;
        v4u a0 = *(const v4u*)(ap);
        v4u a1 = *(const v4u*)(ap + 8);
        v4u b0 = *(const v4u*)(bp);
        v4u b1 = *(const v4u*)(bp + 8);
        BF16x16 a, b;
        a.u = __builtin_shufflevector(a0, a1, 0, 1, 2, 3, 4, 5, 6, 7);
        b.u = __builtin_shufflevector(b0, b1, 0, 1, 2, 3, 4, 5, 6, 7);
        acc = __builtin_amdgcn_wmma_f32_16x16x32_bf16(false, a.h, false, b.h,
                                                      (short)0, acc, false, false);
    }
    float bi = bias[ocb + nm];
    int obase = (tile * 16) * 64 + ocb + nm;
    #pragma unroll
    for (int r = 0; r < 8; ++r)
        h1[obase + (r + 8 * half) * 64] = fmaxf(acc[r] + bi, 0.f);
}

// lin2 (64x64, no relu): tiny, f32 VALU
__global__ void k_lin2(const float* __restrict__ h1, const float* __restrict__ w,
                       const float* __restrict__ b, float* __restrict__ xg)
{
    int e = blockIdx.x * blockDim.x + threadIdx.x;
    if (e >= 128 * 64) return;
    int oc = e & 63, row = e >> 6;
    float acc = b[oc];
    for (int k = 0; k < 64; ++k) acc += h1[row * 64 + k] * w[k * 64 + oc];
    xg[e] = acc;
}

// GCN normalization matrix: D^{-1/2} (A+I) D^{-1/2}, per graph
__global__ void k_norm(const float* __restrict__ adj, float* __restrict__ norm)
{
    __shared__ float ah[256];
    __shared__ float dinv[16];
    int b = blockIdx.x, t = threadIdx.x;
    int i = t >> 4, j = t & 15;
    ah[t] = adj[b * 256 + t] + (i == j ? 1.f : 0.f);
    __syncthreads();
    if (t < 16) {
        float s = 0.f;
        for (int jj = 0; jj < 16; ++jj) s += ah[t * 16 + jj];
        dinv[t] = (s > 0.f) ? rsqrtf(s) : 0.f;
    }
    __syncthreads();
    norm[b * 256 + t] = dinv[i] * ah[t] * dinv[j];
}

// One GCN layer: relu(norm @ (x @ W) + b), per graph (16x64, tiny -> f32 VALU)
// Tiles staged with CDNA5 async global->LDS copies.
__global__ void k_gcn(const float* __restrict__ xin, const float* __restrict__ w,
                      const float* __restrict__ bias, const float* __restrict__ norm,
                      float* __restrict__ xout)
{
    __shared__ __align__(16) float xs[1024], tt[1024], nmat[256];
    int b = blockIdx.x, t = threadIdx.x;
    async_b128(&xs[4 * t], &xin[b * 1024 + 4 * t]);          // 256 thr x 16B = 4KB
    if (t < 64) async_b128(&nmat[4 * t], &norm[b * 256 + 4 * t]);
    async_wait0();
    __syncthreads();
    for (int e = t; e < 1024; e += 256) {
        int i = e >> 6, d = e & 63;
        float a = 0.f;
        for (int k = 0; k < 64; ++k) a += xs[i * 64 + k] * w[k * 64 + d];
        tt[e] = a;
    }
    __syncthreads();
    for (int e = t; e < 1024; e += 256) {
        int i = e >> 6, d = e & 63;
        float a = bias[d];
        for (int j = 0; j < 16; ++j) a += nmat[i * 16 + j] * tt[j * 64 + d];
        xout[b * 1024 + e] = fmaxf(a, 0.f);
    }
}

// Output head: relu(xg @ out1 + b1) @ out2 + b2 -> [8][16][10]
__global__ void k_head(const float* __restrict__ xg, const float* __restrict__ w1,
                       const float* __restrict__ b1, const float* __restrict__ w2,
                       const float* __restrict__ b2, float* __restrict__ out)
{
    __shared__ __align__(16) float xs[1024], tt[1024];
    int b = blockIdx.x, t = threadIdx.x;
    async_b128(&xs[4 * t], &xg[b * 1024 + 4 * t]);
    async_wait0();
    __syncthreads();
    for (int e = t; e < 1024; e += 256) {
        int i = e >> 6, d = e & 63;
        float a = b1[d];
        for (int k = 0; k < 64; ++k) a += xs[i * 64 + k] * w1[k * 64 + d];
        tt[e] = fmaxf(a, 0.f);
    }
    __syncthreads();
    if (t < 160) {
        int i = t / 10, o = t % 10;
        float a = b2[o];
        for (int k = 0; k < 64; ++k) a += tt[i * 64 + k] * w2[k * 10 + o];
        out[b * 160 + t] = a;
    }
}

// ---------------------------------------------------------------------------
extern "C" void kernel_launch(void* const* d_in, const int* in_sizes, int n_in,
                              void* d_out, int out_size, void* d_ws, size_t ws_size,
                              hipStream_t stream)
{
    // dict order: x, adj, conv_ws[0..5], conv_bs[0..5], ln_g[0..2], ln_b[0..2],
    //             lin1_w, lin1_b, lin2_w, lin2_b, gcn_ws[0..2], gcn_bs[0..2],
    //             out1_w, out1_b, out2_w, out2_b
    const float* x    = (const float*)d_in[0];
    const float* adj  = (const float*)d_in[1];
    const float* cw[6]; for (int i = 0; i < 6; ++i) cw[i] = (const float*)d_in[2 + i];
    const float* cb[6]; for (int i = 0; i < 6; ++i) cb[i] = (const float*)d_in[8 + i];
    const float* lng[3]; for (int i = 0; i < 3; ++i) lng[i] = (const float*)d_in[14 + i];
    const float* lnb[3]; for (int i = 0; i < 3; ++i) lnb[i] = (const float*)d_in[17 + i];
    const float* l1w = (const float*)d_in[20]; const float* l1b = (const float*)d_in[21];
    const float* l2w = (const float*)d_in[22]; const float* l2b = (const float*)d_in[23];
    const float* gw[3]; for (int i = 0; i < 3; ++i) gw[i] = (const float*)d_in[24 + i];
    const float* gb[3]; for (int i = 0; i < 3; ++i) gb[i] = (const float*)d_in[27 + i];
    const float* o1w = (const float*)d_in[30]; const float* o1b = (const float*)d_in[31];
    const float* o2w = (const float*)d_in[32]; const float* o2b = (const float*)d_in[33];
    (void)in_sizes; (void)n_in; (void)out_size; (void)ws_size;

    char* ws = (char*)d_ws;
    size_t off = 0;
    auto nxt = [&](size_t bytes) { size_t o = off; off += (bytes + 255) & ~(size_t)255; return o; };
    float*          A0   = (float*)(ws + nxt(33554432));          // f32 NHWC conv out (ping)
    float*          A1   = (float*)(ws + nxt(33554432));          // f32 NHWC conv out (pong)
    unsigned short* P0   = (unsigned short*)(ws + nxt(18939904)); // bf16 padded act (ping)
    unsigned short* P1   = (unsigned short*)(ws + nxt(18939904)); // bf16 padded act (pong)
    unsigned short* WTB  = (unsigned short*)(ws + nxt(368640));   // 5 convs bf16 [tap][oc][ic]
    unsigned short* L1B  = (unsigned short*)(ws + nxt(524288));   // lin1_w bf16 [oc][k]
    unsigned short* FEAT = (unsigned short*)(ws + nxt(1048576));  // bf16 [128][4096] CHW
    float*          MV   = (float*)(ws + nxt(1024));
    float*          H1   = (float*)(ws + nxt(32768));
    float*          XG0  = (float*)(ws + nxt(32768));
    float*          XG1  = (float*)(ws + nxt(32768));
    float*          NORM = (float*)(ws + nxt(8192));

    // weight prep
    k_prep_convw<<<(5 * 36864) / 256, 256, 0, stream>>>(cw[1], cw[2], cw[3], cw[4], cw[5], WTB);
    k_prep_lin1<<<(64 * 4096) / 256, 256, 0, stream>>>(l1w, L1B);

    // layer 0: conv0+pool+relu -> P0(34^2); conv2 wmma 32x32 -> A0; LN -> P1
    k_conv0_pool_pad<<<(128 * 34 * 34) / 256, 256, 0, stream>>>(x, cw[0], cb[0], P0);
    k_conv_wmma<<<dim3(64, 128), 128, 0, stream>>>(P0, WTB + 0 * 36864, cb[1], A0, 5);
    k_ln_stats<<<128, 256, 0, stream>>>(A0, MV, 1024);
    k_ln_relu_pad<<<(128 * 34 * 34 * 64 + 255) / 256, 256, 0, stream>>>(A0, MV, lng[0], lnb[0], P1, 32, 32);

    // layer 1: conv wmma 32x32 -> A1; pool -> P0(18^2); conv wmma 16x16 -> A0; LN -> P1
    k_conv_wmma<<<dim3(64, 128), 128, 0, stream>>>(P1, WTB + 1 * 36864, cb[2], A1, 5);
    k_pool_relu_pad<<<(128 * 18 * 18 * 64 + 255) / 256, 256, 0, stream>>>(A1, P0, 32, 32);
    k_conv_wmma<<<dim3(16, 128), 128, 0, stream>>>(P0, WTB + 2 * 36864, cb[3], A0, 4);
    k_ln_stats<<<128, 256, 0, stream>>>(A0, MV, 256);
    k_ln_relu_pad<<<(128 * 18 * 18 * 64 + 255) / 256, 256, 0, stream>>>(A0, MV, lng[1], lnb[1], P1, 16, 16);

    // layer 2: conv wmma 16x16 -> A1; pool -> P0(10^2); conv wmma 8x8 -> A0; LN -> FEAT
    k_conv_wmma<<<dim3(16, 128), 128, 0, stream>>>(P1, WTB + 3 * 36864, cb[4], A1, 4);
    k_pool_relu_pad<<<(128 * 10 * 10 * 64 + 255) / 256, 256, 0, stream>>>(A1, P0, 16, 16);
    k_conv_wmma<<<dim3(4, 128), 128, 0, stream>>>(P0, WTB + 4 * 36864, cb[5], A0, 3);
    k_ln_stats<<<128, 256, 0, stream>>>(A0, MV, 64);
    k_ln_relu_flat<<<(128 * 4096) / 256, 256, 0, stream>>>(A0, MV, lng[2], lnb[2], FEAT);

    // FC head into graph features
    k_lin1_wmma<<<8, 128, 0, stream>>>(FEAT, L1B, l1b, H1);
    k_lin2<<<32, 256, 0, stream>>>(H1, l2w, l2b, XG0);

    // GCN + output head
    k_norm<<<8, 256, 0, stream>>>(adj, NORM);
    k_gcn<<<8, 256, 0, stream>>>(XG0, gw[0], gb[0], NORM, XG1);
    k_gcn<<<8, 256, 0, stream>>>(XG1, gw[1], gb[1], NORM, XG0);
    k_gcn<<<8, 256, 0, stream>>>(XG0, gw[2], gb[2], NORM, XG1);
    k_head<<<8, 256, 0, stream>>>(XG1, o1w, o1b, o2w, o2b, (float*)d_out);
}
